// DKGCN_26852135534607
// MI455X (gfx1250) — matrix operand order repaired and checked by reading, not compiled, not run on hardware
//
#include <hip/hip_runtime.h>
#include <hip/hip_bf16.h>

// ---------------------------------------------------------------------------
// DKGCN (adaptive graph-conv GRU) for MI455X / gfx1250, wave32 + WMMA bf16.
//
// Restructured algorithm:
//   A = softmax(relu(E E^T))                       (bf16, 2 MB, lives in L2)
//   Per-node weights W[n] = sum_d E[n,d] Wp[d]     (bf16, generated once)
//   Per step/layer:
//     cat  = [x_t, h, A x_t, A h]        -> gate GEMM -> z,r  (sigmoid)
//     ccat = [x_t, z*h, A x_t, A (z*h)]  -> cand GEMM -> hc   (tanh)
//     h    = r*h + (1-r)*hc
//   A@x_t parts are recurrence-free: precomputed for all T in one GEMM.
// All GEMMs: v_wmma_f32_16x16x32_bf16, 2x4 16x16 tiles per wave (operand
// reuse: 8 WMMA per 12 b128 loads), f32 accumulate.
// ---------------------------------------------------------------------------

typedef __attribute__((ext_vector_type(16))) __bf16 v16bf;
typedef __attribute__((ext_vector_type(8)))  __bf16 v8bf;
typedef __attribute__((ext_vector_type(8)))  float  v8f;

#define NN    1024   // nodes
#define BB    32     // batch
#define TT    12     // time steps
#define HIDD  64
#define PP    256    // padded K*Ci (both layers; zero-padded weights)

// ---- WMMA helpers (operand layouts per CDNA5 ISA 7.12.2) -------------------

__device__ __forceinline__ v8f wmma_bf16(v16bf a, v16bf b, v8f c) {
  return __builtin_amdgcn_wmma_f32_16x16x32_bf16(false, a, false, b,
                                                 (short)0, c, false, false);
}

// A-operand 16x32 tile: base points at (rowTile, kChunk) of a row-major
// [rows][ld] bf16 matrix.  lane&15 = row M, lane>>4 selects K-halves.
__device__ __forceinline__ v16bf load_a_tile(const __bf16* base, int ld, int lane) {
  const __bf16* p = base + (size_t)(lane & 15) * ld + ((lane >> 4) << 3);
  union { v16bf v; v8bf h[2]; } u;
  u.h[0] = *(const v8bf*)(p);        // K = koff .. koff+7
  u.h[1] = *(const v8bf*)(p + 16);   // K = 16+koff .. 16+koff+7
  return u.v;
}

// B-operand 32x16 tile: source stored K-contiguous, i.e. row-major
// [cols][ldK]; base points at (colTile, kChunk). lane&15 = col N,
// lane>>4 selects K-halves: K = koff..koff+7 and 16+koff..16+koff+7.
__device__ __forceinline__ v16bf load_b_tile(const __bf16* base, int ld, int lane) {
  const __bf16* p = base + (size_t)(lane & 15) * ld + ((lane >> 4) << 3);
  union { v16bf v; v8bf h[2]; } u;
  u.h[0] = *(const v8bf*)(p);
  u.h[1] = *(const v8bf*)(p + 16);
  return u.v;
}

// ---- adjacency: A = softmax(relu(E E^T)) row-wise, stored bf16 -------------

__global__ void k_build_A(const float* __restrict__ E, __bf16* __restrict__ A) {
  __shared__ float red[256];
  __shared__ float en[16];
  const int n = blockIdx.x, tid = threadIdx.x;
  if (tid < 16) en[tid] = E[n * 16 + tid];
  __syncthreads();
  float v[4];
  float mx = 0.f;                       // relu output >= 0
  for (int j = 0; j < 4; ++j) {
    const int m = tid * 4 + j;
    float s = 0.f;
    #pragma unroll
    for (int d = 0; d < 16; ++d) s += en[d] * E[m * 16 + d];
    s = s > 0.f ? s : 0.f;
    v[j] = s;
    mx = fmaxf(mx, s);
  }
  red[tid] = mx; __syncthreads();
  for (int off = 128; off; off >>= 1) {
    if (tid < off) red[tid] = fmaxf(red[tid], red[tid + off]);
    __syncthreads();
  }
  mx = red[0]; __syncthreads();
  float sm = 0.f;
  for (int j = 0; j < 4; ++j) { v[j] = __expf(v[j] - mx); sm += v[j]; }
  red[tid] = sm; __syncthreads();
  for (int off = 128; off; off >>= 1) {
    if (tid < off) red[tid] += red[tid + off];
    __syncthreads();
  }
  const float inv = 1.f / red[0];
  for (int j = 0; j < 4; ++j)
    A[(size_t)n * NN + tid * 4 + j] = (__bf16)(v[j] * inv);
}

// ---- per-node weight generation: Wt[n][o][i] = sum_d E[n,d] Wp[d,k,ii,o] ---
// i = k*Ci + ii packed; i >= 2*Ci zero-padded up to PP.  K-contiguous layout
// so node GEMMs can load it as a WMMA B operand directly.

__global__ void k_gen_weights(const float* __restrict__ E, const float* __restrict__ Wp,
                              __bf16* __restrict__ Wt, int Ci, int Co, int total) {
  const int idx = blockIdx.x * 256 + threadIdx.x;
  if (idx >= total) return;
  const int i = idx & (PP - 1);
  const int o = (idx >> 8) % Co;
  const int n = idx / (PP * Co);
  float acc = 0.f;
  if (i < 2 * Ci) {
    const int k = i / Ci, ii = i - k * Ci;
    const float* wp = Wp + ((size_t)(k * Ci + ii) * Co + o);
    const size_t dstride = (size_t)2 * Ci * Co;
    #pragma unroll
    for (int d = 0; d < 16; ++d) acc += E[n * 16 + d] * wp[d * dstride];
  }
  Wt[idx] = (__bf16)acc;
}

__global__ void k_gen_bias(const float* __restrict__ E, const float* __restrict__ bp,
                           float* __restrict__ bias, int Co, int total) {
  const int idx = blockIdx.x * 256 + threadIdx.x;
  if (idx >= total) return;
  const int o = idx % Co, n = idx / Co;
  float acc = 0.f;
  #pragma unroll
  for (int d = 0; d < 16; ++d) acc += E[n * 16 + d] * bp[d * Co + o];
  bias[idx] = acc;
}

// ---- aggregation GEMM: Y[n][c] = sum_m A[n][m] * Xt[c][m] ------------------
// One wave per 32x64 output tile (2 row tiles x 4 col tiles: both operands
// reused -> 8 WMMA per 12 b128 loads per K-chunk).
// Output address: n*pitchN + (c>>6)*bStride + cOff + (c&63), which lets us
// scatter results straight into the packed [n][b][feature] cat buffers
// (bStride = row pitch per b), or store linearly with bStride = 64.

__global__ void k_agg(const __bf16* __restrict__ A, const __bf16* __restrict__ Xt,
                      __bf16* __restrict__ Y, int ldXt,
                      int pitchN, int bStride, int cOff) {
  const int lane = threadIdx.x & 31;
  const int rt = blockIdx.x;            // node tile (32 rows)
  const int cg = blockIdx.y;            // column group (64 cols)
  v8f acc[2][4] = {};
  const __bf16* Ab = A + (size_t)rt * 32 * NN;
  const __bf16* Xb = Xt + (size_t)cg * 64 * ldXt;
  for (int kc = 0; kc < NN / 32; ++kc) {
    const v16bf a0 = load_a_tile(Ab + kc * 32, NN, lane);
    const v16bf a1 = load_a_tile(Ab + (size_t)16 * NN + kc * 32, NN, lane);
    // stream next A chunk toward the caches (global_prefetch_b8)
    __builtin_prefetch(Ab + (size_t)(lane & 15) * NN + (kc + 1) * 32, 0, 0);
    #pragma unroll
    for (int j = 0; j < 4; ++j) {
      const v16bf b = load_b_tile(Xb + (size_t)j * 16 * ldXt + kc * 32, ldXt, lane);
      acc[0][j] = wmma_bf16(a0, b, acc[0][j]);
      acc[1][j] = wmma_bf16(a1, b, acc[1][j]);
    }
  }
  const int col = lane & 15, rhi = (lane >> 4) << 3;
  #pragma unroll
  for (int rt2 = 0; rt2 < 2; ++rt2) {
    #pragma unroll
    for (int j = 0; j < 4; ++j) {
      const int cbase = cg * 64 + j * 16;
      __bf16* yp = Y + (size_t)(rt * 32 + rt2 * 16) * pitchN
                     + (size_t)(cbase >> 6) * bStride + cOff + (cbase & 63) + col;
      #pragma unroll
      for (int r = 0; r < 8; ++r)
        yp[(size_t)(rhi + r) * pitchN] = (__bf16)acc[rt2][j][r];
    }
  }
}

// ---- per-node GEMM + GRU epilogue ------------------------------------------
// One block per node n; each wave owns 64 output columns (2x4 WMMA tiles).
// Gate (Co=128): 2 waves -> wave0 = z path, wave1 = r path (no divergence).
// Cand (Co=64) : 1 wave.
//   mode 0 (gate): zr = sigmoid(cat @ W + b); z -> u=z*h (into Ccat and
//                  transposed uT for the next aggregation); r -> Rbuf.
//   mode 1 (cand): hc = tanh(...); h = r*h + (1-r)*hc, plus bf16 copies
//                  ([n][b][f] and transposed [c][n]).

__global__ void k_node_gemm(const __bf16* __restrict__ cat, const __bf16* __restrict__ Wt,
                            const float* __restrict__ bias, int Co, int mode,
                            const float* __restrict__ h_in,
                            __bf16* __restrict__ uCat, int uOff,
                            __bf16* __restrict__ uT,
                            float* __restrict__ Rbuf,
                            float* __restrict__ h_out,
                            __bf16* __restrict__ hCopy,
                            __bf16* __restrict__ hT) {
  const int n = blockIdx.x;
  const int lane = threadIdx.x & 31;
  const int wv = threadIdx.x >> 5;                 // column group of 64
  const __bf16* catB = cat + (size_t)n * BB * PP;
  const __bf16* wB   = Wt + (size_t)n * Co * PP + (size_t)(wv * 64) * PP;
  v8f acc[2][4] = {};
  for (int kc = 0; kc < PP / 32; ++kc) {
    const v16bf a0 = load_a_tile(catB + kc * 32, PP, lane);
    const v16bf a1 = load_a_tile(catB + (size_t)16 * PP + kc * 32, PP, lane);
    // weights are the big streamed operand: prefetch the next K chunk
    __builtin_prefetch(wB + (size_t)(lane & 15) * PP + (kc + 1) * 32, 0, 0);
    #pragma unroll
    for (int j = 0; j < 4; ++j) {
      const v16bf b = load_b_tile(wB + (size_t)j * 16 * PP + kc * 32, PP, lane);
      acc[0][j] = wmma_bf16(a0, b, acc[0][j]);
      acc[1][j] = wmma_bf16(a1, b, acc[1][j]);
    }
  }
  const int col = lane & 15, rhi = (lane >> 4) << 3;
  #pragma unroll
  for (int rt2 = 0; rt2 < 2; ++rt2) {
    #pragma unroll
    for (int ctl = 0; ctl < 4; ++ctl) {
      const int o = wv * 64 + ctl * 16 + col;
      const float bval = bias[(size_t)n * Co + o];
      #pragma unroll
      for (int r = 0; r < 8; ++r) {
        const int b = rt2 * 16 + rhi + r;
        const float v = acc[rt2][ctl][r] + bval;
        if (mode == 0) {                       // gate: sigmoid -> z | r
          const float s = 1.f / (1.f + __expf(-v));
          if (o < HIDD) {
            const float u = s * h_in[((size_t)n * BB + b) * HIDD + o];
            uCat[((size_t)n * BB + b) * PP + uOff + o] = (__bf16)u;
            uT[(size_t)(b * HIDD + o) * NN + n] = (__bf16)u;
          } else {
            Rbuf[((size_t)n * BB + b) * HIDD + (o - HIDD)] = s;
          }
        } else {                               // candidate + state update
          const float hc = tanhf(v);
          const size_t ix = ((size_t)n * BB + b) * HIDD + o;
          const float rr = Rbuf[ix];
          const float hn = rr * h_out[ix] + (1.f - rr) * hc;
          h_out[ix] = hn;
          if (hCopy) hCopy[ix] = (__bf16)hn;
          hT[(size_t)(b * HIDD + o) * NN + n] = (__bf16)hn;
        }
      }
    }
  }
}

// ---- pack x / h / A@x slices into the cat buffers --------------------------
// cat layout per (n,b): [x(Cx) | h-or-u(64) | Ax(Cx) | Ah-or-Au(64) | pad]

__global__ void k_pack(int layer, int t,
                       const float*  __restrict__ x,      // layer1 raw x [B,T,N,1]
                       const __bf16* __restrict__ xprev,  // layer2: h1[t] slice [N][B][64]
                       const __bf16* __restrict__ Axpre,  // layer1: [N][384]; layer2: [N][24576]
                       const float*  __restrict__ h,      // f32 state [N][B][64]
                       __bf16* __restrict__ G, __bf16* __restrict__ C,
                       int Cx, int total) {
  const int idx = blockIdx.x * 256 + threadIdx.x;
  if (idx >= total) return;
  const int span = 2 * Cx + HIDD;
  const int j = idx % span;
  const int b = (idx / span) & (BB - 1);
  const int n = idx / (span * BB);
  const size_t nb = ((size_t)n * BB + b) * PP;
  const int Ci = Cx + HIDD;
  if (j < Cx) {
    __bf16 v;
    if (layer == 1) v = (__bf16)x[((size_t)(b * TT + t)) * NN + n];
    else            v = xprev[((size_t)n * BB + b) * HIDD + j];
    G[nb + j] = v; C[nb + j] = v;
  } else if (j < Cx + HIDD) {
    const int f = j - Cx;
    G[nb + Cx + f] = (__bf16)h[((size_t)n * BB + b) * HIDD + f];
  } else {
    const int jj = j - Cx - HIDD;
    __bf16 v;
    if (layer == 1) v = Axpre[(size_t)n * (TT * BB) + t * BB + b];
    else            v = Axpre[(size_t)n * (TT * BB * HIDD) + t * (BB * HIDD) + b * HIDD + jj];
    G[nb + Ci + jj] = v; C[nb + Ci + jj] = v;
  }
}

// xT[c=(t*32+b)][m=n] bf16 from x f32 — B-operand source for A @ x_all.
__global__ void k_axpack(const float* __restrict__ x, __bf16* __restrict__ xT, int total) {
  const int idx = blockIdx.x * 256 + threadIdx.x;
  if (idx >= total) return;
  const int n = idx & (NN - 1);
  const int c = idx >> 10;
  const int t = c >> 5, b = c & 31;
  xT[idx] = (__bf16)x[((size_t)(b * TT + t)) * NN + n];
}

__global__ void k_zero32(unsigned int* __restrict__ p, int count) {
  const int i = blockIdx.x * 256 + threadIdx.x;
  if (i < count) p[i] = 0u;
}

// out[b][o][n] = bout[o] + sum_f h2[n][b][f] * Wout[o][f]
__global__ void k_out(const float* __restrict__ h2, const float* __restrict__ Wout,
                      const float* __restrict__ bout, float* __restrict__ out) {
  const int idx = blockIdx.x * 256 + threadIdx.x;   // B*12*N
  const int n = idx & (NN - 1);
  const int o = (idx >> 10) % 12;
  const int b = idx / (12 * NN);
  float acc = bout[o];
  const float* hp = h2 + ((size_t)n * BB + b) * HIDD;
  #pragma unroll
  for (int f = 0; f < HIDD; ++f) acc += hp[f] * Wout[o * HIDD + f];
  out[idx] = acc;
}

// ---------------------------------------------------------------------------

extern "C" void kernel_launch(void* const* d_in, const int* in_sizes, int n_in,
                              void* d_out, int out_size, void* d_ws, size_t ws_size,
                              hipStream_t stream) {
  const float* x    = (const float*)d_in[0];
  const float* E    = (const float*)d_in[1];
  const float* Wg1  = (const float*)d_in[2];
  const float* bg1  = (const float*)d_in[3];
  const float* Wc1  = (const float*)d_in[4];
  const float* bc1  = (const float*)d_in[5];
  const float* Wg2  = (const float*)d_in[6];
  const float* bg2  = (const float*)d_in[7];
  const float* Wc2  = (const float*)d_in[8];
  const float* bc2  = (const float*)d_in[9];
  const float* Wout = (const float*)d_in[10];
  const float* bout = (const float*)d_in[11];
  float* out = (float*)d_out;

  // workspace carve-out (~420 MB)
  char* ws = (char*)d_ws;
  size_t off = 0;
  auto alloc = [&](size_t bytes) -> char* {
    char* p = ws + off; off += (bytes + 255) & ~(size_t)255; return p;
  };
  __bf16* A      = (__bf16*)alloc((size_t)NN * NN * 2);
  __bf16* Wt1g   = (__bf16*)alloc((size_t)NN * 128 * PP * 2);
  __bf16* Wt1c   = (__bf16*)alloc((size_t)NN * 64 * PP * 2);
  __bf16* Wt2g   = (__bf16*)alloc((size_t)NN * 128 * PP * 2);
  __bf16* Wt2c   = (__bf16*)alloc((size_t)NN * 64 * PP * 2);
  float*  B1g    = (float*)alloc((size_t)NN * 128 * 4);
  float*  B1c    = (float*)alloc((size_t)NN * 64 * 4);
  float*  B2g    = (float*)alloc((size_t)NN * 128 * 4);
  float*  B2c    = (float*)alloc((size_t)NN * 64 * 4);
  __bf16* xT     = (__bf16*)alloc((size_t)TT * BB * NN * 2);
  __bf16* Ax1    = (__bf16*)alloc((size_t)NN * TT * BB * 2);
  __bf16* h1all  = (__bf16*)alloc((size_t)TT * NN * BB * HIDD * 2);
  __bf16* h1Tall = (__bf16*)alloc((size_t)TT * BB * HIDD * NN * 2);
  __bf16* Ah1    = (__bf16*)alloc((size_t)NN * TT * BB * HIDD * 2);
  __bf16* Gcat   = (__bf16*)alloc((size_t)NN * BB * PP * 2);
  __bf16* Ccat   = (__bf16*)alloc((size_t)NN * BB * PP * 2);
  float*  h      = (float*)alloc((size_t)NN * BB * HIDD * 4);
  __bf16* hT     = (__bf16*)alloc((size_t)BB * HIDD * NN * 2);
  __bf16* uT     = (__bf16*)alloc((size_t)BB * HIDD * NN * 2);
  float*  Rbuf   = (float*)alloc((size_t)NN * BB * HIDD * 4);

  auto blocks = [](int total) { return (total + 255) / 256; };

  // --- one-shot precompute ---
  k_build_A<<<NN, 256, 0, stream>>>(E, A);
  k_gen_weights<<<blocks(NN * 128 * PP), 256, 0, stream>>>(E, Wg1, Wt1g, 65, 128, NN * 128 * PP);
  k_gen_weights<<<blocks(NN * 64  * PP), 256, 0, stream>>>(E, Wc1, Wt1c, 65,  64, NN * 64  * PP);
  k_gen_weights<<<blocks(NN * 128 * PP), 256, 0, stream>>>(E, Wg2, Wt2g, 128, 128, NN * 128 * PP);
  k_gen_weights<<<blocks(NN * 64  * PP), 256, 0, stream>>>(E, Wc2, Wt2c, 128,  64, NN * 64  * PP);
  k_gen_bias<<<blocks(NN * 128), 256, 0, stream>>>(E, bg1, B1g, 128, NN * 128);
  k_gen_bias<<<blocks(NN * 64),  256, 0, stream>>>(E, bc1, B1c, 64,  NN * 64);
  k_gen_bias<<<blocks(NN * 128), 256, 0, stream>>>(E, bg2, B2g, 128, NN * 128);
  k_gen_bias<<<blocks(NN * 64),  256, 0, stream>>>(E, bc2, B2c, 64,  NN * 64);

  // A @ x for all (t,b) at once:  Ax1[n][t*32+b]
  k_axpack<<<blocks(TT * BB * NN), 256, 0, stream>>>(x, xT, TT * BB * NN);
  k_agg<<<dim3(NN / 32, (TT * BB) / 64), 32, 0, stream>>>(A, xT, Ax1, NN, TT * BB, 64, 0);

  // --- layer 1 (Cx = 1, Ci = 65, Ah/Au offset = 66) ---
  k_zero32<<<blocks(NN * BB * HIDD), 256, 0, stream>>>((unsigned int*)h, NN * BB * HIDD);
  k_zero32<<<blocks(BB * HIDD * NN / 2), 256, 0, stream>>>((unsigned int*)hT, BB * HIDD * NN / 2);
  for (int t = 0; t < TT; ++t) {
    const __bf16* hTprev = (t == 0) ? hT : (h1Tall + (size_t)(t - 1) * BB * HIDD * NN);
    k_pack<<<blocks(NN * BB * 66), 256, 0, stream>>>(1, t, x, nullptr, Ax1, h, Gcat, Ccat, 1, NN * BB * 66);
    k_agg<<<dim3(NN / 32, (BB * HIDD) / 64), 32, 0, stream>>>(A, hTprev, Gcat, NN, BB * PP, PP, 66);
    k_node_gemm<<<NN, 64, 0, stream>>>(Gcat, Wt1g, B1g, 128, 0,
                                       h, Ccat, 1, uT, Rbuf, nullptr, nullptr, nullptr);
    k_agg<<<dim3(NN / 32, (BB * HIDD) / 64), 32, 0, stream>>>(A, uT, Ccat, NN, BB * PP, PP, 66);
    k_node_gemm<<<NN, 32, 0, stream>>>(Ccat, Wt1c, B1c, 64, 1,
                                       nullptr, nullptr, 0, nullptr, Rbuf, h,
                                       h1all  + (size_t)t * NN * BB * HIDD,
                                       h1Tall + (size_t)t * BB * HIDD * NN);
  }

  // A @ h1[t] for all t in one GEMM: Ah1[n][t*2048 + b*64 + f]
  k_agg<<<dim3(NN / 32, (TT * BB * HIDD) / 64), 32, 0, stream>>>(A, h1Tall, Ah1, NN,
                                                                TT * BB * HIDD, 64, 0);

  // --- layer 2 (Cx = 64, Ci = 128, Ah/Au offset = 192) ---
  k_zero32<<<blocks(NN * BB * HIDD), 256, 0, stream>>>((unsigned int*)h, NN * BB * HIDD);
  k_zero32<<<blocks(BB * HIDD * NN / 2), 256, 0, stream>>>((unsigned int*)hT, BB * HIDD * NN / 2);
  for (int t = 0; t < TT; ++t) {
    k_pack<<<blocks(NN * BB * 192), 256, 0, stream>>>(2, t, nullptr,
                                                      h1all + (size_t)t * NN * BB * HIDD,
                                                      Ah1, h, Gcat, Ccat, 64, NN * BB * 192);
    k_agg<<<dim3(NN / 32, (BB * HIDD) / 64), 32, 0, stream>>>(A, hT, Gcat, NN, BB * PP, PP, 192);
    k_node_gemm<<<NN, 64, 0, stream>>>(Gcat, Wt2g, B2g, 128, 0,
                                       h, Ccat, 64, uT, Rbuf, nullptr, nullptr, nullptr);
    k_agg<<<dim3(NN / 32, (BB * HIDD) / 64), 32, 0, stream>>>(A, uT, Ccat, NN, BB * PP, PP, 192);
    k_node_gemm<<<NN, 32, 0, stream>>>(Ccat, Wt2c, B2c, 64, 1,
                                       nullptr, nullptr, 0, nullptr, Rbuf, h, nullptr, hT);
  }

  // --- output head: [B, HOR=12, N, 1] ---
  k_out<<<blocks(BB * 12 * NN), 256, 0, stream>>>(h, Wout, bout, out);
}
